// RadarRnn3_24558622999002
// MI455X (gfx1250) — matrix operand order repaired and checked by reading, not compiled, not run on hardware
//
#include <hip/hip_runtime.h>

// ---------------------------------------------------------------------------
// Persistent 16-stage pipelined RNN for MI455X (gfx1250, wave32, WMMA).
//   stage s in [0,8):  stack-1 layer s      (stage 7 also computes fc1 -> mid)
//   stage s in [8,16): stack-2 layer s-8    (stage 15 also computes out -> y)
// One workgroup (256 thr = 8 waves) per stage; per-step global barrier.
// Stage kind is template-specialized so the recurrence loop has no uniform
// branches: C starts from a bias-preloaded fragment, two chained
// v_wmma_f32_16x16x32_f16 per C tile, tanh epilogue, LDS relayout.
// ---------------------------------------------------------------------------

typedef __attribute__((ext_vector_type(16))) _Float16 v16h;
typedef __attribute__((ext_vector_type(8)))  _Float16 v8h;
typedef __attribute__((ext_vector_type(8)))  float    v8f;

#define DEV __device__ __forceinline__

constexpr int B_      = 256;
constexpr int T_      = 4096;
constexpr int H_      = 32;
constexpr int L_      = 8;
constexpr int NSTAGE  = 16;
constexpr int THREADS = 256;

DEV v16h cat8(v8h lo, v8h hi) {
  return __builtin_shufflevector(lo, hi, 0,1,2,3,4,5,6,7,8,9,10,11,12,13,14,15);
}

// Build a 16x32 f16 A-fragment (wave32 layout: lanes 0-15 rows, K split by
// lane[4]) from a row-major [rows][32] f16 buffer (LDS or global).
DEV v16h loadA(const _Float16* base, int rowBase, int lane) {
  const int row = rowBase + (lane & 15);
  const int k0  = (lane & 16) ? 8 : 0;
  const _Float16* p = base + row * H_ + k0;
  v8h lo = *(const v8h*)(p);
  v8h hi = *(const v8h*)(p + 16);
  return cat8(lo, hi);
}

DEV v8f wmma_f16(v16h a, v16h b, v8f c) {
  return __builtin_amdgcn_wmma_f32_16x16x32_f16(
      /*neg_a=*/false, a, /*neg_b=*/false, b,
      /*c_mod=*/(short)0, c, /*reuse_a=*/false, /*reuse_b=*/false);
}

DEV float fast_tanh(float x) {
  x = fminf(fmaxf(x, -15.0f), 15.0f);
  const float e = __expf(2.0f * x);          // v_exp_f32 path
  return (e - 1.0f) / (e + 1.0f);
}

// Sense-reversing grid barrier over NSTAGE workgroups (bar[0]=count, bar[1]=gen).
DEV void grid_barrier(unsigned* bar) {
  __syncthreads();
  if (threadIdx.x == 0) {
    __threadfence();
    volatile unsigned* genp = bar + 1;
    const unsigned g = *genp;
    const unsigned arrived = atomicAdd(bar, 1u) + 1u;
    if (arrived == (unsigned)NSTAGE) {
      atomicExch(bar, 0u);
      __threadfence();
      atomicAdd(bar + 1, 1u);
    } else {
      while (*genp == g) __builtin_amdgcn_s_sleep(1);
    }
    __threadfence();
  }
  __syncthreads();
}

struct Frags {
  v16h  bwih[2];    // input-projection B fragments (unused for DIM1 stages)
  v16h  bwhh[2];    // recurrent B fragments
  v8f   cbias[2];   // bias-preloaded C fragments (per N tile)
  float xw[2];      // Wih0 column weight (DIM1 stages)
};

// Fully specialized stage loop: DIM1 = scalar-input first layer of a stack,
// PRODUCER = writes f16 ring for next stage, STK2 = stack-2 stage.
template <bool DIM1, bool PRODUCER, bool STK2>
DEV void run_stage(int l, int tid, int lane, int wave,
                   _Float16* hbuf, float* xs, const float* dotw, float dotb,
                   const Frags fr,
                   const float* __restrict__ x,
                   const float* __restrict__ h0_in,
                   float* __restrict__ yout,
                   unsigned* __restrict__ bar,
                   float* __restrict__ mid,
                   const _Float16* __restrict__ ring_in,
                   _Float16* __restrict__ ring_out,
                   float* __restrict__ h1buf)
{
  const int mbase = wave * 32;               // this wave's batch rows
  const int roff0 = (lane & 16) ? 8 : 0;     // D-layout row offset within tile
  const int col0  = lane & 15;               // D-layout column within N tile
  const int start = STK2 ? (T_ + l) : l;     // first global step for this stage

  v16h ah[2] = {};                           // hidden-state A fragments

  for (int step = 0; step < 2 * T_ + L_; ++step) {
    grid_barrier(bar);
    const int u = step - start;              // local timestep
    if (u < 0 || u >= T_) continue;

    // ---- initialize hidden at u==0 (stack2 reads stack1's published h1) ----
    if (u == 0) {
      const float* h0 = (STK2 ? h1buf : h0_in) + l * B_ * H_;
      #pragma unroll
      for (int i = tid; i < B_ * H_; i += THREADS) hbuf[i] = (_Float16)h0[i];
      __syncthreads();
      #pragma unroll
      for (int m = 0; m < 2; ++m) ah[m] = loadA(hbuf, mbase + 16 * m, lane);
    }

    // ---- fetch this step's input ----
    v16h ain[2];
    if (DIM1) {
      if (STK2) {
        xs[tid] = mid[(size_t)u * B_ + tid];
      } else {
        xs[tid] = x[(size_t)tid * T_ + u];
        if (u + 16 < T_) __builtin_prefetch(&x[(size_t)tid * T_ + u + 16], 0, 3);
      }
    } else {
      const _Float16* rin = ring_in + (u & 1) * (B_ * H_);
      #pragma unroll
      for (int m = 0; m < 2; ++m) ain[m] = loadA(rin, mbase + 16 * m, lane);
    }
    __syncthreads();   // xs visible; all reads of previous hbuf complete

    // DIM1: gather this lane's 16 scalar inputs up-front (pipelined ds loads)
    float xr[16];
    if (DIM1) {
      #pragma unroll
      for (int m = 0; m < 2; ++m)
        #pragma unroll
        for (int j = 0; j < 8; ++j)
          xr[m * 8 + j] = xs[mbase + 16 * m + roff0 + j];
    }

    // ---- GEMMs: C = bias (+ x*w) + A_in*Wih^T + h*Whh^T ----
    v8f c[2][2];
    #pragma unroll
    for (int m = 0; m < 2; ++m) {
      #pragma unroll
      for (int n = 0; n < 2; ++n) {
        if (DIM1) {
          v8f c0;
          #pragma unroll
          for (int j = 0; j < 8; ++j)
            c0[j] = fr.cbias[n][j] + xr[m * 8 + j] * fr.xw[n];
          c[m][n] = wmma_f16(ah[m], fr.bwhh[n], c0);
        } else {
          v8f acc = wmma_f16(ain[m], fr.bwih[n], fr.cbias[n]);
          c[m][n] = wmma_f16(ah[m], fr.bwhh[n], acc);
        }
      }
    }

    // ---- epilogue: tanh only (bias already folded into C) ----
    float hv[2][2][8];
    #pragma unroll
    for (int m = 0; m < 2; ++m)
      #pragma unroll
      for (int n = 0; n < 2; ++n)
        #pragma unroll
        for (int j = 0; j < 8; ++j)
          hv[m][n][j] = fast_tanh(c[m][n][j]);

    // ---- store new hidden to LDS (D-layout -> row-major relayout) ----
    #pragma unroll
    for (int m = 0; m < 2; ++m) {
      const int rb = mbase + 16 * m + roff0;
      #pragma unroll
      for (int n = 0; n < 2; ++n) {
        const int col = n * 16 + col0;
        #pragma unroll
        for (int j = 0; j < 8; ++j)
          hbuf[(rb + j) * H_ + col] = (_Float16)hv[m][n][j];
      }
    }

    // ---- final-step f32 state publication (off the hot path) ----
    if (u == T_ - 1) {
      #pragma unroll
      for (int m = 0; m < 2; ++m) {
        const int rb = mbase + 16 * m + roff0;
        #pragma unroll
        for (int n = 0; n < 2; ++n) {
          const int col = n * 16 + col0;
          #pragma unroll
          for (int j = 0; j < 8; ++j) {
            const size_t idx = (size_t)l * B_ * H_ + (size_t)(rb + j) * H_ + col;
            if (STK2) yout[(size_t)B_ * T_ + idx] = hv[m][n][j];   // h2 out
            else      h1buf[idx]                  = hv[m][n][j];   // -> stack2
          }
        }
      }
    }
    __syncthreads();

    // ---- reload hidden A-fragments for the next step ----
    #pragma unroll
    for (int m = 0; m < 2; ++m) ah[m] = loadA(hbuf, mbase + 16 * m, lane);

    // ---- publish: ring copy, or fused Linear(32,1) epilogue ----
    if (PRODUCER) {
      const uint4* srcp = (const uint4*)hbuf;
      uint4* dstp = (uint4*)(ring_out + (u & 1) * (B_ * H_));
      #pragma unroll
      for (int i = tid; i < (B_ * H_ * 2) / 16; i += THREADS) dstp[i] = srcp[i];
    } else {
      float acc = dotb;
      const _Float16* hr = hbuf + tid * H_;
      #pragma unroll
      for (int k = 0; k < H_; ++k) acc += (float)hr[k] * dotw[k];
      if (!STK2) mid[(size_t)u * B_ + tid] = acc;                  // -> stack2
      else       yout[(size_t)tid * T_ + u] = acc;                 // final y
    }
  }
}

__global__ __launch_bounds__(THREADS)
void radar_rnn3_pipeline(
    const float* __restrict__ x,        // [B,T] (I==1)
    const float* __restrict__ h0_in,    // [L,B,H]
    const float* __restrict__ w_ih0_1,  // [H]
    const float* __restrict__ w_ih_1,   // [L-1,H,H]
    const float* __restrict__ w_hh_1,   // [L,H,H]
    const float* __restrict__ b_ih_1,   // [L,H]
    const float* __restrict__ b_hh_1,   // [L,H]
    const float* __restrict__ fc1_w,    // [H]
    const float* __restrict__ fc1_b,    // [1]
    const float* __restrict__ w_ih0_2,
    const float* __restrict__ w_ih_2,
    const float* __restrict__ w_hh_2,
    const float* __restrict__ b_ih_2,
    const float* __restrict__ b_hh_2,
    const float* __restrict__ out_w,    // [H]
    const float* __restrict__ out_b,    // [1]
    float* __restrict__ yout,           // d_out: [B,T] y  then [L,B,H] h2
    unsigned* __restrict__ bar,         // ws: barrier words
    float* __restrict__ mid,            // ws: [T,B] stage7 -> stage8 hand-off
    _Float16* __restrict__ ring,        // ws: [16][2][B][H] activation rings
    float* __restrict__ h1buf)          // ws: [L][B][H] stack1 final hidden
{
  const int s     = blockIdx.x;
  const int tid   = threadIdx.x;
  const int lane  = tid & 31;
  const int wave  = tid >> 5;
  const bool stk2 = (s >= 8);
  const int  l    = stk2 ? (s - 8) : s;      // layer index within its stack
  const bool dim1 = (l == 0);                // scalar input layer

  __shared__ __align__(32) _Float16 hbuf[B_ * H_];   // current hidden, f16
  __shared__ __align__(32) _Float16 wih_s[H_ * H_];
  __shared__ __align__(32) _Float16 whh_s[H_ * H_];
  __shared__ float xs[B_];                           // scalar input per batch
  __shared__ float wv0[H_];                          // Wih0 column weights
  __shared__ float dotw[H_];                         // fc1 / out weights
  __shared__ float biasc[H_];                        // bih + bhh per column

  // ---- one-time weight prep (f32 -> f16 B-layout source rows in LDS) ----
  const float* Whh = (stk2 ? w_hh_2 : w_hh_1) + l * H_ * H_;
  #pragma unroll
  for (int i = tid; i < H_ * H_; i += THREADS) whh_s[i] = (_Float16)Whh[i];
  if (!dim1) {
    const float* Wih = (stk2 ? w_ih_2 : w_ih_1) + (l - 1) * H_ * H_;
    #pragma unroll
    for (int i = tid; i < H_ * H_; i += THREADS) wih_s[i] = (_Float16)Wih[i];
  } else if (tid < H_) {
    wv0[tid] = (stk2 ? w_ih0_2 : w_ih0_1)[tid];
  }
  if (tid < H_) {
    biasc[tid] = (stk2 ? b_ih_2 : b_ih_1)[l * H_ + tid] +
                 (stk2 ? b_hh_2 : b_hh_1)[l * H_ + tid];
    if (s == 7)  dotw[tid] = fc1_w[tid];
    if (s == 15) dotw[tid] = out_w[tid];
  }
  __syncthreads();

  // ---- per-lane cached register fragments ----
  Frags fr;
  #pragma unroll
  for (int n = 0; n < 2; ++n) {
    const int ng = n * 16 + (lane & 15);       // output column
    const int kb = (lane & 16) ? 8 : 0;        // K offset (k = kb..kb+7, 16+kb..16+kb+7)
    fr.bwhh[n] = cat8(*(const v8h*)(&whh_s[ng * H_ + kb]), *(const v8h*)(&whh_s[ng * H_ + kb + 16]));
    fr.bwih[n] = dim1 ? fr.bwhh[n] : cat8(*(const v8h*)(&wih_s[ng * H_ + kb]), *(const v8h*)(&wih_s[ng * H_ + kb + 16]));
    const float b = biasc[ng];
    #pragma unroll
    for (int j = 0; j < 8; ++j) fr.cbias[n][j] = b;  // same col -> same bias
    fr.xw[n] = dim1 ? wv0[ng] : 0.0f;
  }
  const float dotb = (s == 7) ? fc1_b[0] : ((s == 15) ? out_b[0] : 0.0f);

  const _Float16* ring_in  = ring + (size_t)(s - 1) * (2 * B_ * H_);
  _Float16*       ring_out = ring + (size_t)s       * (2 * B_ * H_);

  // ---- dispatch to the fully specialized stage loop ----
  if (s == 0) {
    run_stage<true , true , false>(l, tid, lane, wave, hbuf, xs, dotw, dotb, fr,
                                   x, h0_in, yout, bar, mid, ring_in, ring_out, h1buf);
  } else if (s == 7) {
    run_stage<false, false, false>(l, tid, lane, wave, hbuf, xs, dotw, dotb, fr,
                                   x, h0_in, yout, bar, mid, ring_in, ring_out, h1buf);
  } else if (s < 8) {
    run_stage<false, true , false>(l, tid, lane, wave, hbuf, xs, dotw, dotb, fr,
                                   x, h0_in, yout, bar, mid, ring_in, ring_out, h1buf);
  } else if (s == 8) {
    run_stage<true , true , true >(l, tid, lane, wave, hbuf, xs, dotw, dotb, fr,
                                   x, h0_in, yout, bar, mid, ring_in, ring_out, h1buf);
  } else if (s == 15) {
    run_stage<false, false, true >(l, tid, lane, wave, hbuf, xs, dotw, dotb, fr,
                                   x, h0_in, yout, bar, mid, ring_in, ring_out, h1buf);
  } else {
    run_stage<false, true , true >(l, tid, lane, wave, hbuf, xs, dotw, dotb, fr,
                                   x, h0_in, yout, bar, mid, ring_in, ring_out, h1buf);
  }
}

__global__ void radar_rnn3_init(unsigned* bar) {
  if (threadIdx.x < 64) bar[threadIdx.x] = 0u;
}

extern "C" void kernel_launch(void* const* d_in, const int* in_sizes, int n_in,
                              void* d_out, int out_size, void* d_ws, size_t ws_size,
                              hipStream_t stream) {
  const float* x       = (const float*)d_in[0];
  const float* h_state = (const float*)d_in[1];
  const float* w_ih0_1 = (const float*)d_in[2];
  const float* w_ih_1  = (const float*)d_in[3];
  const float* w_hh_1  = (const float*)d_in[4];
  const float* b_ih_1  = (const float*)d_in[5];
  const float* b_hh_1  = (const float*)d_in[6];
  const float* fc1_w   = (const float*)d_in[7];
  const float* fc1_b   = (const float*)d_in[8];
  const float* w_ih0_2 = (const float*)d_in[9];
  const float* w_ih_2  = (const float*)d_in[10];
  const float* w_hh_2  = (const float*)d_in[11];
  const float* b_ih_2  = (const float*)d_in[12];
  const float* b_hh_2  = (const float*)d_in[13];
  const float* out_w   = (const float*)d_in[14];
  const float* out_b   = (const float*)d_in[15];
  float* y = (float*)d_out;

  char* ws = (char*)d_ws;
  size_t off = 0;
  unsigned* bar = (unsigned*)(ws + off);        off += 256;
  float*    mid = (float*)(ws + off);           off += (size_t)T_ * B_ * sizeof(float);
  _Float16* ring = (_Float16*)(ws + off);       off += (size_t)NSTAGE * 2 * B_ * H_ * sizeof(_Float16);
  float*    h1b = (float*)(ws + off);           off += (size_t)L_ * B_ * H_ * sizeof(float);

  radar_rnn3_init<<<1, 64, 0, stream>>>(bar);
  radar_rnn3_pipeline<<<NSTAGE, THREADS, 0, stream>>>(
      x, h_state, w_ih0_1, w_ih_1, w_hh_1, b_ih_1, b_hh_1,
      fc1_w, fc1_b, w_ih0_2, w_ih_2, w_hh_2, b_ih_2, b_hh_2,
      out_w, out_b, y, bar, mid, ring, h1b);
}
